// DiffDecoder_74775380623446
// MI455X (gfx1250) — compile-verified, matrix operand order checked by reading, not run on hardware
//
#include <hip/hip_runtime.h>
#include <cmath>

typedef unsigned short u16;
typedef unsigned int   u32;
typedef __attribute__((ext_vector_type(16))) __bf16 v16bf;
typedef __attribute__((ext_vector_type(8)))  float  v8f;

#define NHEADS 8
#define NFREQ 64
#define PI_F 3.14159265358979f
#define TWO_PI_F 6.28318530717959f

// ---------------- device helpers ----------------
__device__ __forceinline__ u16 f2bf(float f) {
  u32 u = __float_as_uint(f);
  u32 r = u + 0x7fffu + ((u >> 16) & 1u);   // round-to-nearest-even
  return (u16)(r >> 16);
}
__device__ __forceinline__ float bf2f(u16 h) {
  return __uint_as_float(((u32)h) << 16);
}
__device__ __forceinline__ void atomicMaxF(float* a, float v) {
  int old = __float_as_int(*a);
  while (__int_as_float(old) < v) {
    int prev = atomicCAS((int*)a, old, __float_as_int(v));
    if (prev == old) break;
    old = prev;
  }
}

// ---------------- bf16 WMMA GEMM: C = act(A@B + bias [+ Cin]) ----------------
// A: [M,K] bf16 row-major (K multiple of 32, rows 16B aligned)
// B: [K,N] bf16 row-major (N multiple of 8)
// block = 128 threads (4 wave32); block tile (RT*64) x 64.
// Each wave owns RT row-tiles of 16 rows x 64 cols; B fragment loaded once per
// N-tile and reused across the RT row-tiles (amortizes LDS traffic).
// Staging is two-phase (all global loads in flight, then LDS stores) so the
// loads overlap instead of stalling on s_wait_loadcnt per chunk.
template <int RT>
__global__ __launch_bounds__(128) void k_gemm_t(
    const u16* __restrict__ A, const u16* __restrict__ B,
    const float* __restrict__ bias, const float* __restrict__ Cin,
    float* __restrict__ Cout, u16* __restrict__ Cbf,
    int M, int N, int K, int act)
{
  __shared__ u16 As[RT * 64 * 40];  // [row][k], stride 40 u16 (80B: 16B aligned, depitched)
  __shared__ u16 Bs[64 * 40];       // [n][k] (transposed during staging)
  const int tid  = threadIdx.x;
  const int wave = tid >> 5, lane = tid & 31;
  const int half = lane >> 4, lq = lane & 15;
  const int bm = blockIdx.x * (RT * 64), bn = blockIdx.y * 64;
  const bool fullM = (bm + RT * 64 <= M);
  const bool fullN = (bn + 64 <= N);

  // per-thread staging coordinates (compile-time strides; no per-iter muls)
  const int arow0 = tid >> 2;          // 0..31, rows step +32 per chunk
  const int acol  = (tid & 3) << 3;    // 0,8,16,24
  const int kp2   = (tid >> 3) << 1;   // even k: 0,2,..,30 (k-pair base)
  const int nb    = (tid & 7) << 3;    // 0,8,..,56

  v8f z = {0.f,0.f,0.f,0.f,0.f,0.f,0.f,0.f};
  v8f acc[RT][4];
  #pragma unroll
  for (int rt = 0; rt < RT; rt++)
    #pragma unroll
    for (int nt = 0; nt < 4; nt++) acc[rt][nt] = z;

  for (int k0 = 0; k0 < K; k0 += 32) {
    // ---- phase 1: issue ALL global loads for this k-step ----
    uint4 at[RT * 2];
    if (fullM) {
      #pragma unroll
      for (int i = 0; i < RT * 2; i++)
        at[i] = *(const uint4*)(A + (size_t)(bm + arow0 + 32 * i) * K + k0 + acol);
    } else {
      #pragma unroll
      for (int i = 0; i < RT * 2; i++) {
        int gr = bm + arow0 + 32 * i;
        at[i] = make_uint4(0u, 0u, 0u, 0u);
        if (gr < M) at[i] = *(const uint4*)(A + (size_t)gr * K + k0 + acol);
      }
    }
    uint4 b0 = make_uint4(0u, 0u, 0u, 0u), b1 = b0;
    if (fullN || (bn + nb < N)) {
      b0 = *(const uint4*)(B + (size_t)(k0 + kp2)     * N + bn + nb);
      b1 = *(const uint4*)(B + (size_t)(k0 + kp2 + 1) * N + bn + nb);
    }
    if (k0 + 32 < K && tid < RT * 64 && (bm + tid) < M)
      __builtin_prefetch(A + (size_t)(bm + tid) * K + k0 + 32, 0, 0);

    // ---- phase 2: LDS stores (single wait for all loads) ----
    #pragma unroll
    for (int i = 0; i < RT * 2; i++)
      *(uint4*)(&As[(arow0 + 32 * i) * 40 + acol]) = at[i];
    {
      const u16* p0 = (const u16*)&b0;
      const u16* p1 = (const u16*)&b1;
      #pragma unroll
      for (int i = 0; i < 8; i++) {
        u32 v = (u32)p0[i] | ((u32)p1[i] << 16);   // packed k-pair (k, k+1) for col nb+i
        *(u32*)(&Bs[(nb + i) * 40 + kp2]) = v;
      }
    }
    __syncthreads();

    // ---- A fragments (ISA 7.12.2 bf16 16x32 layout) ----
    v16bf afrag[RT];
    #pragma unroll
    for (int rt = 0; rt < RT; rt++) {
      u32* ap = (u32*)&afrag[rt];
      const int arow = (wave * RT + rt) * 16 + lq;
      #pragma unroll
      for (int j = 0; j < 8; j++) {
        int kk = ((j < 4) ? (2 * j) : (16 + 2 * (j - 4))) + 8 * half;
        ap[j] = *(const u32*)(&As[arow * 40 + kk]);
      }
    }
    // ---- per N-tile: one B fragment, RT WMMAs ----
    #pragma unroll
    for (int nt = 0; nt < 4; nt++) {
      v16bf bfrag;
      u32* bp = (u32*)&bfrag;
      const int bcol = nt * 16 + lq;
      #pragma unroll
      for (int j = 0; j < 8; j++) {
        int kk = ((j < 4) ? (2 * j) : (16 + 2 * (j - 4))) + 8 * half;
        bp[j] = *(const u32*)(&Bs[bcol * 40 + kk]);
      }
      #pragma unroll
      for (int rt = 0; rt < RT; rt++)
        acc[rt][nt] = __builtin_amdgcn_wmma_f32_16x16x32_bf16(
            false, afrag[rt], false, bfrag, (short)0, acc[rt][nt], false, false);
    }
    __syncthreads();
  }

  // ---- epilogue: D layout — VGPR i: row = i + 8*half, col = lq ----
  #pragma unroll
  for (int rt = 0; rt < RT; rt++) {
    #pragma unroll
    for (int nt = 0; nt < 4; nt++) {
      int gc = bn + nt * 16 + lq;
      if (gc >= N) continue;
      float bv = bias ? bias[gc] : 0.0f;
      #pragma unroll
      for (int i = 0; i < 8; i++) {
        int gr = bm + (wave * RT + rt) * 16 + half * 8 + i;
        if (gr >= M) continue;
        size_t o = (size_t)gr * N + gc;
        float v = acc[rt][nt][i] + bv;
        if (Cin) v += Cin[o];
        if (act == 1) v = fmaxf(v, 0.0f);
        else if (act == 2) v = 1.0f / (1.0f + __expf(-v));
        if (Cout) Cout[o] = v;
        if (Cbf)  Cbf[o]  = f2bf(v);
      }
    }
  }
}

// ---------------- LayerNorm over 128-wide rows (one wave32 per row) ----------------
__global__ __launch_bounds__(128) void k_ln128(
    const float* __restrict__ x, const float* __restrict__ g, const float* __restrict__ b,
    const float* __restrict__ res, float* __restrict__ y, u16* __restrict__ ybf,
    int n, int relu)
{
  int wave = threadIdx.x >> 5, lane = threadIdx.x & 31;
  int row = blockIdx.x * 4 + wave;
  if (row >= n) return;
  const float4 v4 = ((const float4*)(x + (size_t)row * 128))[lane];
  float vv[4] = {v4.x, v4.y, v4.z, v4.w};
  float s = vv[0] + vv[1] + vv[2] + vv[3];
  #pragma unroll
  for (int o = 16; o; o >>= 1) s += __shfl_xor(s, o, 32);
  float mu = s * (1.0f / 128.0f);
  float q = 0.f;
  #pragma unroll
  for (int c = 0; c < 4; c++) { float d = vv[c] - mu; q += d * d; }
  #pragma unroll
  for (int o = 16; o; o >>= 1) q += __shfl_xor(q, o, 32);
  float inv = rsqrtf(q * (1.0f / 128.0f) + 1e-5f);
  #pragma unroll
  for (int c = 0; c < 4; c++) {
    int idx = lane * 4 + c;
    float o = (vv[c] - mu) * inv * g[idx] + b[idx];
    if (relu) o = fmaxf(o, 0.0f);
    if (res)  o += res[(size_t)row * 128 + idx];
    if (y)    y[(size_t)row * 128 + idx]   = o;
    if (ybf)  ybf[(size_t)row * 128 + idx] = f2bf(o);
  }
}

// ---------------- small elementwise kernels ----------------
__global__ void k_cvt_pad_rows(const float* __restrict__ src, u16* __restrict__ dst,
                               int M, int K, int Kp) {
  int t = blockIdx.x * 256 + threadIdx.x;
  if (t >= M * Kp) return;
  int m = t / Kp, k = t - m * Kp;
  dst[t] = (k < K) ? f2bf(src[(size_t)m * K + k]) : (u16)0;
}
__global__ void k_cvt_pad_w(const float* __restrict__ src, u16* __restrict__ dst,
                            int K, int Kp, int N) {
  int t = blockIdx.x * 256 + threadIdx.x;
  if (t >= Kp * N) return;
  int k = t / N, nn = t - k * N;
  dst[t] = (k < K) ? f2bf(src[(size_t)k * N + nn]) : (u16)0;
}
__global__ void k_fill(float* p, float v, int n) {
  int t = blockIdx.x * 256 + threadIdx.x;
  if (t < n) p[t] = v;
}
__global__ void k_build_rel(const float* __restrict__ pos_s, const float* __restrict__ head_s,
                            const float* __restrict__ pos_m, const float* __restrict__ head_m,
                            const int* __restrict__ src, const int* __restrict__ dst,
                            int E, int din, int is_t2m, float* __restrict__ rel) {
  int e = blockIdx.x * 256 + threadIdx.x;
  if (e >= E) return;
  int s = src[e], d = dst[e];
  float rx = pos_s[2 * s]     - pos_m[2 * d];
  float ry = pos_s[2 * s + 1] - pos_m[2 * d + 1];
  float dist = sqrtf(rx * rx + ry * ry);
  float hx = cosf(head_m[d]), hy = sinf(head_m[d]);
  float ang = atan2f(hx * ry - hy * rx, hx * rx + hy * ry);
  float dh = head_s[s] - head_m[d];
  dh = fmodf(dh + PI_F, TWO_PI_F);
  if (dh < 0.f) dh += TWO_PI_F;
  dh -= PI_F;
  rel[(size_t)e * din]     = dist;
  rel[(size_t)e * din + 1] = ang;
  rel[(size_t)e * din + 2] = dh;
  if (is_t2m) rel[(size_t)e * din + 3] = (float)(s % 11) - 10.0f;
}
// fourier features for one dim d: F[e, 0..63]=cos, [64..127]=sin, [128]=x, [129..159]=0 (pad)
__global__ void k_feat(const float* __restrict__ rel, const float* __restrict__ freqs,
                       int E, int din, int d, u16* __restrict__ F) {
  int t = blockIdx.x * 256 + threadIdx.x;
  if (t >= E * 64) return;
  int e = t >> 6, f = t & 63;
  float x = rel[(size_t)e * din + d];
  float ang = x * freqs[f] * TWO_PI_F;
  F[(size_t)e * 160 + f]      = f2bf(cosf(ang));
  F[(size_t)e * 160 + 64 + f] = f2bf(sinf(ang));
  if (f == 0) {
    F[(size_t)e * 160 + 128] = f2bf(x);
    for (int p = 129; p < 160; p++) F[(size_t)e * 160 + p] = 0;
  }
}
__global__ void k_score(const float* __restrict__ q, const u16* __restrict__ kall,
                        const u16* __restrict__ kr, const int* __restrict__ src,
                        const int* __restrict__ dst, int E, float* __restrict__ sim) {
  int t = blockIdx.x * 256 + threadIdx.x;
  if (t >= E * NHEADS) return;
  int e = t >> 3, h = t & 7;
  int s = src[e], d = dst[e];
  const float* qp = q    + (size_t)d * 128 + h * 16;
  const u16*   kp = kall + (size_t)s * 128 + h * 16;
  const u16*   rp = kr   + (size_t)e * 128 + h * 16;
  float acc = 0.f;
  #pragma unroll
  for (int j = 0; j < 16; j++) acc += qp[j] * (bf2f(kp[j]) + bf2f(rp[j]));
  sim[t] = acc * 0.25f;   // HEADDIM^-0.5
}
__global__ void k_segmax(const float* __restrict__ sim, const int* __restrict__ dst,
                         int E, float* __restrict__ smax) {
  int t = blockIdx.x * 256 + threadIdx.x;
  if (t >= E * NHEADS) return;
  int e = t >> 3, h = t & 7;
  atomicMaxF(&smax[dst[e] * 8 + h], sim[t]);
}
__global__ void k_expsum(const float* __restrict__ sim, const float* __restrict__ smax,
                         const int* __restrict__ dst, int E,
                         float* __restrict__ ex, float* __restrict__ den) {
  int t = blockIdx.x * 256 + threadIdx.x;
  if (t >= E * NHEADS) return;
  int e = t >> 3, h = t & 7;
  float sm = smax[dst[e] * 8 + h];
  if (!(sm > -1e37f)) sm = 0.f;     // isfinite() replacement for -inf init
  float v = __expf(sim[t] - sm);
  ex[t] = v;
  atomicAdd(&den[dst[e] * 8 + h], v);
}
__global__ void k_agg(const float* __restrict__ ex, const float* __restrict__ den,
                      const u16* __restrict__ vall, const u16* __restrict__ vr,
                      const int* __restrict__ src, const int* __restrict__ dst,
                      int E, float* __restrict__ agg) {
  int t = blockIdx.x * 256 + threadIdx.x;
  if (t >= E * NHEADS) return;
  int e = t >> 3, h = t & 7;
  int s = src[e], d = dst[e];
  float coef = ex[t] / (den[d * 8 + h] + 1e-16f);
  const u16* vp = vall + (size_t)s * 128 + h * 16;
  const u16* rp = vr   + (size_t)e * 128 + h * 16;
  float* apg = agg + (size_t)d * 128 + h * 16;
  #pragma unroll
  for (int j = 0; j < 16; j++) atomicAdd(&apg[j], (bf2f(vp[j]) + bf2f(rp[j])) * coef);
}
__global__ void k_concat(const float* __restrict__ agg, const float* __restrict__ xd,
                         int A_, u16* __restrict__ cat) {
  int t = blockIdx.x * 256 + threadIdx.x;
  if (t >= A_ * 256) return;
  int r = t >> 8, c = t & 255;
  float v = (c < 128) ? agg[(size_t)r * 128 + c] : xd[(size_t)r * 128 + c - 128];
  cat[t] = f2bf(v);
}
__global__ void k_gate(const float* __restrict__ agg, const float* __restrict__ g,
                       const float* __restrict__ s, int n, u16* __restrict__ upd) {
  int t = blockIdx.x * 256 + threadIdx.x;
  if (t >= n) return;
  float a = agg[t];
  upd[t] = f2bf(a + g[t] * (s[t] - a));
}
__global__ void k_scatter(const float* __restrict__ src, float* __restrict__ dst,
                          int A_, int w, int off, int stride) {
  int t = blockIdx.x * 256 + threadIdx.x;
  if (t >= A_ * w) return;
  int r = t / w, c = t - r * w;
  dst[(size_t)r * stride + off + c] = src[t];
}

// ================= host side =================
struct Bump {
  char* p;
  template <class T> T* take(size_t count) {
    T* r = (T*)p;
    p += ((count * sizeof(T)) + 255) & ~(size_t)255;
    return r;
  }
};

// pytree leaves in sorted-dict-key order (jax flatten order)
struct MlpP  { const float *b1,*b2,*be1,*g1,*w1,*w2; };
struct FourP { const float *b1,*b2,*be1,*be2,*bo,*freqs,*g1,*g2,*w1,*w2,*wo; };
struct AttnP { const float *bdst,*bf1,*bf2,*bff,*bffp,*bg,*bo,*bpost,*bq,*br,*bs,*bsrc,
               *gdst,*gff,*gffp,*gpost,*gr,*gsrc,*wf1,*wf2,*wg,*wk,*wkr,*wo,*wq,*ws,*wv,*wvr; };

static void loadMlp(MlpP& m, void* const* d, int& i) {
  const float** f = (const float**)&m;
  for (int k = 0; k < 6; k++) f[k] = (const float*)d[i++];
}
static void loadFour(FourP& m, void* const* d, int& i) {
  const float** f = (const float**)&m;
  for (int k = 0; k < 11; k++) f[k] = (const float*)d[i++];
}
static void loadAttn(AttnP& m, void* const* d, int& i) {
  const float** f = (const float**)&m;
  for (int k = 0; k < 28; k++) f[k] = (const float*)d[i++];
}

struct AttnW { u16 *wq,*wk,*wv,*wkr,*wvr,*ws,*wg,*wo,*wf1,*wf2; };
struct FourW { u16* w1[4]; u16* w2[4]; u16* wo; int din; };

static void cvtw(hipStream_t st, const float* src, u16* dst, int K, int Kp, int N) {
  k_cvt_pad_w<<<(Kp * N + 255) / 256, 256, 0, st>>>(src, dst, K, Kp, N);
}
static void gemm(hipStream_t st, const u16* Ab, const u16* Bb, const float* bias,
                 const float* Cin, float* Cout, u16* Cbf, int M, int N, int K, int act) {
  if (M >= 4096) {   // big edge GEMMs: 128x64 tile, B fragments amortized over 2 row tiles
    dim3 g((M + 127) / 128, (N + 63) / 64);
    k_gemm_t<2><<<g, 128, 0, st>>>(Ab, Bb, bias, Cin, Cout, Cbf, M, N, K, act);
  } else {           // small node GEMMs: 64x64 tile for occupancy
    dim3 g((M + 63) / 64, (N + 63) / 64);
    k_gemm_t<1><<<g, 128, 0, st>>>(Ab, Bb, bias, Cin, Cout, Cbf, M, N, K, act);
  }
}
static void ln(hipStream_t st, const float* x, const float* g_, const float* b,
               const float* res, float* y, u16* ybf, int n, int relu) {
  k_ln128<<<(n + 3) / 4, 128, 0, st>>>(x, g_, b, res, y, ybf, n, relu);
}
static void fill(hipStream_t st, float* p, float v, int n) {
  k_fill<<<(n + 255) / 256, 256, 0, st>>>(p, v, n);
}

static void prepAttn(hipStream_t st, const AttnP& p, AttnW& w, Bump& bp) {
  w.wq  = bp.take<u16>(128 * 128); cvtw(st, p.wq,  w.wq,  128, 128, 128);
  w.wk  = bp.take<u16>(128 * 128); cvtw(st, p.wk,  w.wk,  128, 128, 128);
  w.wv  = bp.take<u16>(128 * 128); cvtw(st, p.wv,  w.wv,  128, 128, 128);
  w.wkr = bp.take<u16>(128 * 128); cvtw(st, p.wkr, w.wkr, 128, 128, 128);
  w.wvr = bp.take<u16>(128 * 128); cvtw(st, p.wvr, w.wvr, 128, 128, 128);
  w.ws  = bp.take<u16>(128 * 128); cvtw(st, p.ws,  w.ws,  128, 128, 128);
  w.wg  = bp.take<u16>(256 * 128); cvtw(st, p.wg,  w.wg,  256, 256, 128);
  w.wo  = bp.take<u16>(128 * 128); cvtw(st, p.wo,  w.wo,  128, 128, 128);
  w.wf1 = bp.take<u16>(128 * 512); cvtw(st, p.wf1, w.wf1, 128, 128, 512);
  w.wf2 = bp.take<u16>(512 * 128); cvtw(st, p.wf2, w.wf2, 512, 512, 128);
}
static void prepFour(hipStream_t st, const FourP& p, FourW& w, int din, Bump& bp) {
  w.din = din;
  for (int d = 0; d < din; d++) {
    w.w1[d] = bp.take<u16>(160 * 128);
    cvtw(st, p.w1 + (size_t)d * 129 * 128, w.w1[d], 129, 160, 128);
    w.w2[d] = bp.take<u16>(128 * 128);
    cvtw(st, p.w2 + (size_t)d * 128 * 128, w.w2[d], 128, 128, 128);
  }
  w.wo = bp.take<u16>(128 * 128);
  cvtw(st, p.wo, w.wo, 128, 128, 128);
}

struct Scratch {
  float *m, *xd, *qf, *sim, *ex, *smax, *den, *agg, *gf, *sf, *h2f, *Hf, *S, *rel;
  float *r_t2m, *r_pl, *r_a, *r_m, *tmp80;
  u16 *xdbf, *xsbf, *kallbf, *vallbf, *rnbf, *krbf, *vrbf, *catbf, *updbf;
  u16 *h1bf, *hbf, *Hbf, *F, *ngbf, *mbf;
};

// fourier encoder: out = relu-stack per dim then wo projection (optionally += Cin alias out)
static void run_fourier(hipStream_t st, const FourP& p, const FourW& w,
                        const float* rel, int E, Scratch& sc,
                        const float* Cin, float* outF32) {
  fill(st, sc.S, 0.0f, E * 128);
  for (int d = 0; d < w.din; d++) {
    k_feat<<<(E * 64 + 255) / 256, 256, 0, st>>>(rel, p.freqs + d * NFREQ, E, w.din, d, sc.F);
    gemm(st, sc.F, w.w1[d], p.b1 + d * 128, nullptr, sc.Hf, nullptr, E, 128, 160, 0);
    ln(st, sc.Hf, p.g1 + d * 128, p.be1 + d * 128, nullptr, nullptr, sc.Hbf, E, 1);
    gemm(st, sc.Hbf, w.w2[d], p.b2 + d * 128, sc.S, sc.S, nullptr, E, 128, 128, 0);
  }
  ln(st, sc.S, p.g2, p.be2, nullptr, nullptr, sc.Hbf, E, 1);
  gemm(st, sc.Hbf, w.wo, p.bo, Cin, outF32, nullptr, E, 128, 128, 0);
}

static void run_attn(hipStream_t st, const AttnP& p, const AttnW& w,
                     const float* x_src, int Ns, const float* r,
                     const int* src, const int* dst, int E, int A_, Scratch& sc) {
  ln(st, x_src, p.gsrc, p.bsrc, nullptr, nullptr, sc.xsbf, Ns, 0);
  ln(st, sc.m,  p.gdst, p.bdst, nullptr, sc.xd,   sc.xdbf, A_, 0);
  ln(st, r,     p.gr,   p.br,   nullptr, nullptr, sc.rnbf, E,  0);
  gemm(st, sc.xdbf, w.wq,  p.bq,   nullptr, sc.qf,  nullptr,   A_, 128, 128, 0);
  gemm(st, sc.xsbf, w.wk,  nullptr, nullptr, nullptr, sc.kallbf, Ns, 128, 128, 0);
  gemm(st, sc.xsbf, w.wv,  nullptr, nullptr, nullptr, sc.vallbf, Ns, 128, 128, 0);
  gemm(st, sc.rnbf, w.wkr, nullptr, nullptr, nullptr, sc.krbf,   E,  128, 128, 0);
  gemm(st, sc.rnbf, w.wvr, nullptr, nullptr, nullptr, sc.vrbf,   E,  128, 128, 0);
  fill(st, sc.smax, -3.0e38f, A_ * 8);
  fill(st, sc.den,  0.0f,     A_ * 8);
  fill(st, sc.agg,  0.0f,     A_ * 128);
  int EH = E * NHEADS;
  k_score <<<(EH + 255) / 256, 256, 0, st>>>(sc.qf, sc.kallbf, sc.krbf, src, dst, E, sc.sim);
  k_segmax<<<(EH + 255) / 256, 256, 0, st>>>(sc.sim, dst, E, sc.smax);
  k_expsum<<<(EH + 255) / 256, 256, 0, st>>>(sc.sim, sc.smax, dst, E, sc.ex, sc.den);
  k_agg   <<<(EH + 255) / 256, 256, 0, st>>>(sc.ex, sc.den, sc.vallbf, sc.vrbf, src, dst, E, sc.agg);
  k_concat<<<(A_ * 256 + 255) / 256, 256, 0, st>>>(sc.agg, sc.xd, A_, sc.catbf);
  gemm(st, sc.catbf, w.wg, p.bg, nullptr, sc.gf, nullptr, A_, 128, 256, 2);
  gemm(st, sc.xdbf,  w.ws, p.bs, nullptr, sc.sf, nullptr, A_, 128, 128, 0);
  k_gate<<<(A_ * 128 + 255) / 256, 256, 0, st>>>(sc.agg, sc.gf, sc.sf, A_ * 128, sc.updbf);
  gemm(st, sc.updbf, w.wo, p.bo, nullptr, sc.h2f, nullptr, A_, 128, 128, 0);
  ln(st, sc.h2f, p.gpost, p.bpost, sc.m, sc.m, nullptr, A_, 0);
  // FFN
  ln(st, sc.m, p.gff, p.bff, nullptr, nullptr, sc.hbf, A_, 0);
  gemm(st, sc.hbf,  w.wf1, p.bf1, nullptr, nullptr, sc.h1bf, A_, 512, 128, 1);
  gemm(st, sc.h1bf, w.wf2, p.bf2, nullptr, sc.h2f, nullptr,  A_, 128, 512, 0);
  ln(st, sc.h2f, p.gffp, p.bffp, sc.m, sc.m, nullptr, A_, 0);
}

extern "C" void kernel_launch(void* const* d_in, const int* in_sizes, int n_in,
                              void* d_out_v, int out_size, void* d_ws, size_t ws_size,
                              hipStream_t stream) {
  const float* noised_gt    = (const float*)d_in[0];
  const float* noise_labels = (const float*)d_in[1];
  const float* pos_m  = (const float*)d_in[2];
  const float* head_m = (const float*)d_in[3];
  const float* pos_t  = (const float*)d_in[4];
  const float* head_t = (const float*)d_in[5];
  const float* pos_pl = (const float*)d_in[6];
  const float* head_pl= (const float*)d_in[7];
  const float* x_t    = (const float*)d_in[8];
  const float* x_pl   = (const float*)d_in[9];
  const float* x_a    = (const float*)d_in[10];
  float* d_out = (float*)d_out_v;

  const int A_ = in_sizes[1];
  const int Nt = in_sizes[5];
  const int Pn = in_sizes[7];
  const int* e_t2m = (const int*)d_in[n_in - 4]; int Et  = in_sizes[n_in - 4] / 2;
  const int* e_pl  = (const int*)d_in[n_in - 3]; int Epl = in_sizes[n_in - 3] / 2;
  const int* e_a   = (const int*)d_in[n_in - 2]; int Ea  = in_sizes[n_in - 2] / 2;
  const int* e_m   = (const int*)d_in[n_in - 1]; int Em  = in_sizes[n_in - 1] / 2;

  // params pytree (sorted top-level keys): a2m, m2m, mlp_in, mlp_out, noise_emb,
  // pl2m, r_a2m, r_m2m, r_pl2m, r_t2m, t2m
  int pi = 11;
  AttnP a2m_p[2];  loadAttn(a2m_p[0], d_in, pi);  loadAttn(a2m_p[1], d_in, pi);
  AttnP m2m_p[2];  loadAttn(m2m_p[0], d_in, pi);  loadAttn(m2m_p[1], d_in, pi);
  MlpP  mlp_in_p;  loadMlp(mlp_in_p, d_in, pi);
  MlpP  mlp_out_p; loadMlp(mlp_out_p, d_in, pi);
  FourP noise_p;   loadFour(noise_p, d_in, pi);
  AttnP pl2m_p[2]; loadAttn(pl2m_p[0], d_in, pi); loadAttn(pl2m_p[1], d_in, pi);
  FourP ra_p;      loadFour(ra_p, d_in, pi);
  FourP rm_p;      loadFour(rm_p, d_in, pi);
  FourP rpl_p;     loadFour(rpl_p, d_in, pi);
  FourP rt_p;      loadFour(rt_p, d_in, pi);
  AttnP t2m_p[2];  loadAttn(t2m_p[0], d_in, pi);  loadAttn(t2m_p[1], d_in, pi);

  Bump bp; bp.p = (char*)d_ws;

  // ---- bf16 weight prep ----
  AttnW t2m_w[2], pl2m_w[2], a2m_w[2], m2m_w[2];
  for (int i = 0; i < 2; i++) {
    prepAttn(stream, t2m_p[i],  t2m_w[i],  bp);
    prepAttn(stream, pl2m_p[i], pl2m_w[i], bp);
    prepAttn(stream, a2m_p[i],  a2m_w[i],  bp);
    prepAttn(stream, m2m_p[i],  m2m_w[i],  bp);
  }
  FourW noise_w, rt_w, rpl_w, ra_w, rm_w;
  prepFour(stream, noise_p, noise_w, 1, bp);
  prepFour(stream, rt_p,    rt_w,    4, bp);
  prepFour(stream, rpl_p,   rpl_w,   3, bp);
  prepFour(stream, ra_p,    ra_w,    3, bp);
  prepFour(stream, rm_p,    rm_w,    3, bp);
  u16* win1 = bp.take<u16>(256 * 128); cvtw(stream, mlp_in_p.w1, win1, 240, 256, 128);
  u16* win2 = bp.take<u16>(128 * 128); cvtw(stream, mlp_in_p.w2, win2, 128, 128, 128);
  u16* wout1 = bp.take<u16>(128 * 128); cvtw(stream, mlp_out_p.w1, wout1, 128, 128, 128);
  u16* wout2 = bp.take<u16>(128 * 80);  cvtw(stream, mlp_out_p.w2, wout2, 128, 128, 80);

  // ---- scratch ----
  int Emax = Et; if (Epl > Emax) Emax = Epl; if (Ea > Emax) Emax = Ea;
  if (Em > Emax) Emax = Em; if (A_ > Emax) Emax = A_;
  int Nmax = Nt; if (Pn > Nmax) Nmax = Pn; if (A_ > Nmax) Nmax = A_;

  Scratch sc;
  sc.m     = bp.take<float>((size_t)A_ * 128);
  sc.xd    = bp.take<float>((size_t)A_ * 128);
  sc.qf    = bp.take<float>((size_t)A_ * 128);
  sc.smax  = bp.take<float>((size_t)A_ * 8);
  sc.den   = bp.take<float>((size_t)A_ * 8);
  sc.agg   = bp.take<float>((size_t)A_ * 128);
  sc.gf    = bp.take<float>((size_t)A_ * 128);
  sc.sf    = bp.take<float>((size_t)A_ * 128);
  sc.h2f   = bp.take<float>((size_t)A_ * 128);
  sc.tmp80 = bp.take<float>((size_t)A_ * 80);
  sc.xdbf  = bp.take<u16>((size_t)A_ * 128);
  sc.catbf = bp.take<u16>((size_t)A_ * 256);
  sc.updbf = bp.take<u16>((size_t)A_ * 128);
  sc.h1bf  = bp.take<u16>((size_t)A_ * 512);
  sc.hbf   = bp.take<u16>((size_t)A_ * 128);
  sc.mbf   = bp.take<u16>((size_t)A_ * 128);
  sc.ngbf  = bp.take<u16>((size_t)A_ * 256);
  sc.xsbf   = bp.take<u16>((size_t)Nmax * 128);
  sc.kallbf = bp.take<u16>((size_t)Nmax * 128);
  sc.vallbf = bp.take<u16>((size_t)Nmax * 128);
  sc.sim  = bp.take<float>((size_t)Emax * 8);
  sc.ex   = bp.take<float>((size_t)Emax * 8);
  sc.rel  = bp.take<float>((size_t)Emax * 4);
  sc.rnbf = bp.take<u16>((size_t)Emax * 128);
  sc.krbf = bp.take<u16>((size_t)Emax * 128);
  sc.vrbf = bp.take<u16>((size_t)Emax * 128);
  sc.F    = bp.take<u16>((size_t)Emax * 160);
  sc.Hbf  = bp.take<u16>((size_t)Emax * 128);
  sc.Hf   = bp.take<float>((size_t)Emax * 128);
  sc.S    = bp.take<float>((size_t)Emax * 128);
  sc.r_t2m = bp.take<float>((size_t)Et * 128);
  sc.r_pl  = bp.take<float>((size_t)Epl * 128);
  sc.r_a   = bp.take<float>((size_t)Ea * 128);
  sc.r_m   = bp.take<float>((size_t)Em * 128);

  // ---- m = mlp_in(noised_gt) + fourier(noise_labels) ----
  k_cvt_pad_rows<<<(A_ * 256 + 255) / 256, 256, 0, stream>>>(noised_gt, sc.ngbf, A_, 240, 256);
  gemm(stream, sc.ngbf, win1, mlp_in_p.b1, nullptr, sc.Hf, nullptr, A_, 128, 256, 0);
  ln(stream, sc.Hf, mlp_in_p.g1, mlp_in_p.be1, nullptr, nullptr, sc.hbf, A_, 1);
  gemm(stream, sc.hbf, win2, mlp_in_p.b2, nullptr, sc.m, nullptr, A_, 128, 128, 0);
  run_fourier(stream, noise_p, noise_w, noise_labels, A_, sc, sc.m, sc.m);

  // ---- relation fourier embeddings ----
  k_build_rel<<<(Et + 255) / 256, 256, 0, stream>>>(pos_t, head_t, pos_m, head_m,
      e_t2m, e_t2m + Et, Et, 4, 1, sc.rel);
  run_fourier(stream, rt_p, rt_w, sc.rel, Et, sc, nullptr, sc.r_t2m);
  k_build_rel<<<(Epl + 255) / 256, 256, 0, stream>>>(pos_pl, head_pl, pos_m, head_m,
      e_pl, e_pl + Epl, Epl, 3, 0, sc.rel);
  run_fourier(stream, rpl_p, rpl_w, sc.rel, Epl, sc, nullptr, sc.r_pl);
  k_build_rel<<<(Ea + 255) / 256, 256, 0, stream>>>(pos_m, head_m, pos_m, head_m,
      e_a, e_a + Ea, Ea, 3, 0, sc.rel);
  run_fourier(stream, ra_p, ra_w, sc.rel, Ea, sc, nullptr, sc.r_a);
  k_build_rel<<<(Em + 255) / 256, 256, 0, stream>>>(pos_m, head_m, pos_m, head_m,
      e_m, e_m + Em, Em, 3, 0, sc.rel);
  run_fourier(stream, rm_p, rm_w, sc.rel, Em, sc, nullptr, sc.r_m);

  // ---- recurrent attention stack ----
  for (int rec = 0; rec < 3; rec++) {
    for (int i = 0; i < 2; i++) {
      run_attn(stream, t2m_p[i],  t2m_w[i],  x_t,  Nt, sc.r_t2m, e_t2m, e_t2m + Et, Et, A_, sc);
      run_attn(stream, pl2m_p[i], pl2m_w[i], x_pl, Pn, sc.r_pl,  e_pl,  e_pl + Epl, Epl, A_, sc);
      run_attn(stream, a2m_p[i],  a2m_w[i],  x_a,  A_, sc.r_a,   e_a,   e_a + Ea,   Ea,  A_, sc);
      run_attn(stream, m2m_p[i],  m2m_w[i],  sc.m, A_, sc.r_m,   e_m,   e_m + Em,   Em,  A_, sc);
    }
    // out_rec = mlp_out(m) -> d_out[:, rec*80 : rec*80+80] (row stride 240)
    k_cvt_pad_rows<<<(A_ * 128 + 255) / 256, 256, 0, stream>>>(sc.m, sc.mbf, A_, 128, 128);
    gemm(stream, sc.mbf, wout1, mlp_out_p.b1, nullptr, sc.Hf, nullptr, A_, 128, 128, 0);
    ln(stream, sc.Hf, mlp_out_p.g1, mlp_out_p.be1, nullptr, nullptr, sc.hbf, A_, 1);
    gemm(stream, sc.hbf, wout2, mlp_out_p.b2, nullptr, sc.tmp80, nullptr, A_, 80, 128, 0);
    k_scatter<<<(A_ * 80 + 255) / 256, 256, 0, stream>>>(sc.tmp80, d_out, A_, 80, rec * 80, 240);
  }
}
